// MapNet_4604204941660
// MI455X (gfx1250) — matrix operand order repaired and checked by reading, not compiled, not run on hardware
//
#include <hip/hip_runtime.h>

#define DCH 128                 // channel dim
#define NMAT 66                 // packed weight matrices
#define MATELEMS (DCH * DCH)    // 16384
#define MATBYTES (MATELEMS * 2) // 32768 bytes packed bf16

typedef __attribute__((ext_vector_type(16))) __bf16 v16bf;
typedef __attribute__((ext_vector_type(8)))  __bf16 v8bf;
typedef __attribute__((ext_vector_type(8)))  float  v8f;

// ---------------------------------------------------------------------------
// Stage one packed 32KB weight matrix into LDS with gfx1250 async DMA.
// 256 threads x 8 x 16B = 32768B. Tracked by ASYNCcnt; each wave waits for its
// own transfers, then the block barrier makes LDS visible to all waves.
// ---------------------------------------------------------------------------
__device__ inline void stage_B_lds(__bf16* Bsh, const __bf16* Bp, int tid)
{
    const unsigned lds0 = (unsigned)(size_t)Bsh + (unsigned)tid * 16u;
    const unsigned va0  = (unsigned)tid * 16u;
    #pragma unroll
    for (int i = 0; i < 8; ++i) {
        asm volatile("global_load_async_to_lds_b128 %0, %1, %2"
                     :
                     : "v"(lds0 + (unsigned)i * 4096u),
                       "v"(va0 + (unsigned)i * 4096u),
                       "s"(Bp)
                     : "memory");
    }
    asm volatile("s_wait_asynccnt 0x0" ::: "memory");
    __syncthreads();
}

// ---------------------------------------------------------------------------
// Pack one 128x128 f32 weight matrix ([K,N] row-major) into WMMA B-fragment
// order, bf16.  frag index = ((kt*8+nt)*32 + lane), 16 bf16 per lane.
// B tile, contiguous k order used here (A matches): lanes 0-15 -> col N=lane, K=e (e=0..15);
// lanes 16-31 -> col N=lane-16, K=16+e.
// ---------------------------------------------------------------------------
__global__ __launch_bounds__(256) void pack_weights(
    const float* __restrict__ w_in2,  const float* __restrict__ w_seg2,
    const float* __restrict__ ctr_w,  const float* __restrict__ ctr2_w,
    const float* __restrict__ left_w, const float* __restrict__ right_w,
    const float* __restrict__ pre_w,  const float* __restrict__ suc_w,
    __bf16* __restrict__ dst)
{
    const int m = blockIdx.y;
    const float* W;
    if      (m == 0)  W = w_in2;
    else if (m == 1)  W = w_seg2;
    else if (m < 6)   W = ctr_w   + (size_t)(m - 2)  * MATELEMS;
    else if (m < 10)  W = ctr2_w  + (size_t)(m - 6)  * MATELEMS;
    else if (m < 14)  W = left_w  + (size_t)(m - 10) * MATELEMS;
    else if (m < 18)  W = right_w + (size_t)(m - 14) * MATELEMS;
    else if (m < 42)  W = pre_w   + (size_t)(m - 18) * MATELEMS;
    else              W = suc_w   + (size_t)(m - 42) * MATELEMS;

    const int idx  = blockIdx.x * 256 + threadIdx.x;   // 0..16383 (grid.x == 64)
    const int j    = idx & 15;
    const int lane = (idx >> 4) & 31;
    const int tile = idx >> 9;                         // kt*8 + nt
    const int nt   = tile & 7;
    const int kt   = tile >> 3;
    const int n    = lane & 15;
    const int kb   = (lane >> 4) << 4;                 // 0 or 16
    const int K    = kt * 32 + kb + j;
    const int Nc   = nt * 16 + n;
    dst[(size_t)m * MATELEMS + idx] = (__bf16)W[(size_t)K * DCH + Nc];
}

// ---------------------------------------------------------------------------
// Core WMMA tile: one wave computes rows m0..m0+15 x all 128 cols.
// A fragments (16-bit A 16x32): lane r=lane&15 -> row; kb=(lane>=16)?16:0.
// B fragments are read from LDS through a 2-deep software pipeline so the
// ds_load pair for fragment f+2 issues before WMMA f consumes fragment f.
// ---------------------------------------------------------------------------
__device__ inline void gemm_wave_tile(
    const __bf16* __restrict__ A, const __bf16* __restrict__ Bsh,
    int m0, int lane, v8f acc[8])
{
    const int r  = lane & 15;
    const int hi = lane >> 4;
    const int kb = hi << 4;

    union AF { v16bf v; struct { v8bf lo, hi; } p; } a[4];
    const __bf16* arow = A + (size_t)(m0 + r) * DCH + kb;
    #pragma unroll
    for (int kt = 0; kt < 4; ++kt) {
        a[kt].p.lo = *(const v8bf*)(arow + kt * 32);
        a[kt].p.hi = *(const v8bf*)(arow + kt * 32 + 8);
    }

    const v16bf* bsh = (const v16bf*)Bsh;
    v16bf bbuf[2];
    bbuf[0] = bsh[lane];                               // frag 0
    bbuf[1] = bsh[32 + lane];                          // frag 1
    #pragma unroll
    for (int f = 0; f < 32; ++f) {
        const v16bf bcur = bbuf[f & 1];
        if (f + 2 < 32)
            bbuf[f & 1] = bsh[(size_t)((f + 2) << 5) + lane];   // prefetch f+2
        const int kt = f >> 3;
        const int nt = f & 7;
        acc[nt] = __builtin_amdgcn_wmma_f32_16x16x32_bf16(
            false, a[kt].v, false, bcur, (short)0, acc[nt], false, false);
    }
}

// C[M,128] = A[M,128](bf16) @ Bpacked, plain f32 store.
__global__ __launch_bounds__(256) void gemm_bf16_128(
    const __bf16* __restrict__ A, const __bf16* __restrict__ Bp,
    float* __restrict__ C, int M)
{
    __shared__ __bf16 Bsh[MATELEMS];
    stage_B_lds(Bsh, Bp, threadIdx.x);

    const int wave = threadIdx.x >> 5;
    const int lane = threadIdx.x & 31;
    const int m0   = blockIdx.x * 128 + wave * 16;
    if (m0 >= M) return;                       // wave-uniform; after barrier

    v8f acc[8];
    #pragma unroll
    for (int nt = 0; nt < 8; ++nt) acc[nt] = (v8f){0.f,0.f,0.f,0.f,0.f,0.f,0.f,0.f};
    gemm_wave_tile(A, Bsh, m0, lane, acc);

    const int col = lane & 15;
    const int mb  = m0 + ((lane >> 4) << 3);
    #pragma unroll
    for (int nt = 0; nt < 8; ++nt) {
        float* crow = C + (size_t)mb * DCH + nt * 16 + col;
        #pragma unroll
        for (int t = 0; t < 8; ++t) crow[(size_t)t * DCH] = acc[nt][t];
    }
}

// Fused: feat = relu(gn(A@Bp, g, b) + feat)  (+ bf16 shadow copy).
// Each wave owns complete rows, so GN row statistics = accumulator sums +
// a 16-lane xor-shuffle reduction (rows live in half-wave groups).
__global__ __launch_bounds__(256) void gemm_bf16_128_gnres(
    const __bf16* __restrict__ A, const __bf16* __restrict__ Bp,
    const float* __restrict__ g, const float* __restrict__ b,
    float* __restrict__ feat, __bf16* __restrict__ featbf, int M)
{
    __shared__ __bf16 Bsh[MATELEMS];
    stage_B_lds(Bsh, Bp, threadIdx.x);

    const int wave = threadIdx.x >> 5;
    const int lane = threadIdx.x & 31;
    const int m0   = blockIdx.x * 128 + wave * 16;
    if (m0 >= M) return;

    v8f acc[8];
    #pragma unroll
    for (int nt = 0; nt < 8; ++nt) acc[nt] = (v8f){0.f,0.f,0.f,0.f,0.f,0.f,0.f,0.f};
    gemm_wave_tile(A, Bsh, m0, lane, acc);

    const int col0 = lane & 15;
    const int mb   = m0 + ((lane >> 4) << 3);

    float gv[8], bv[8];
    #pragma unroll
    for (int nt = 0; nt < 8; ++nt) { gv[nt] = g[nt * 16 + col0]; bv[nt] = b[nt * 16 + col0]; }

    #pragma unroll
    for (int t = 0; t < 8; ++t) {
        float s = 0.f, q = 0.f;
        #pragma unroll
        for (int nt = 0; nt < 8; ++nt) { float x = acc[nt][t]; s += x; q += x * x; }
        #pragma unroll
        for (int off = 8; off > 0; off >>= 1) {      // reduce within 16-lane row group
            s += __shfl_xor(s, off, 32);
            q += __shfl_xor(q, off, 32);
        }
        const float mean = s * (1.f / 128.f);
        const float inv  = rsqrtf(q * (1.f / 128.f) - mean * mean + 1e-5f);
        const size_t ro  = (size_t)(mb + t) * DCH;
        #pragma unroll
        for (int nt = 0; nt < 8; ++nt) {
            const int col = nt * 16 + col0;
            float y = (acc[nt][t] - mean) * inv * gv[nt] + bv[nt] + feat[ro + col];
            y = fmaxf(y, 0.f);
            feat[ro + col]   = y;
            featbf[ro + col] = (__bf16)y;
        }
    }
}

// ---------------------------------------------------------------------------
// Wave-per-row helpers (wave32, 4 channels/lane)
// ---------------------------------------------------------------------------
__device__ inline void wave_reduce2(float& a, float& b) {
    #pragma unroll
    for (int off = 16; off > 0; off >>= 1) {
        a += __shfl_xor(a, off, 32);
        b += __shfl_xor(b, off, 32);
    }
}

__device__ inline void row_stats(const float4& x, float& mean, float& inv) {
    float s = x.x + x.y + x.z + x.w;
    float q = x.x*x.x + x.y*x.y + x.z*x.z + x.w*x.w;
    wave_reduce2(s, q);
    mean = s * (1.f / 128.f);
    float var = q * (1.f / 128.f) - mean * mean;
    inv = rsqrtf(var + 1e-5f);
}

// h = relu(x @ W1[2,128] + b1)  -> bf16
__global__ __launch_bounds__(256) void encoder_hidden(
    const float* __restrict__ x, const float* __restrict__ W1,
    const float* __restrict__ b1, __bf16* __restrict__ out, int Nn)
{
    const int row = blockIdx.x * 8 + (threadIdx.x >> 5);
    if (row >= Nn) return;
    const int lane = threadIdx.x & 31;
    const float c0 = x[(size_t)row * 2 + 0];
    const float c1 = x[(size_t)row * 2 + 1];
    #pragma unroll
    for (int c = 0; c < 4; ++c) {
        const int ch = lane * 4 + c;
        float h = fmaf(c0, W1[ch], fmaf(c1, W1[DCH + ch], b1[ch]));
        out[(size_t)row * DCH + ch] = (__bf16)fmaxf(h, 0.f);
    }
}

// feat = relu(gn(T1,g1,b1) + gn(T2,g2,b2)) -> f32 feat + bf16 copy
__global__ __launch_bounds__(256) void encoder_combine(
    const float* __restrict__ T1, const float* __restrict__ T2,
    const float* __restrict__ g1, const float* __restrict__ b1,
    const float* __restrict__ g2, const float* __restrict__ b2,
    float* __restrict__ feat, __bf16* __restrict__ featbf, int Nn)
{
    const int row = blockIdx.x * 8 + (threadIdx.x >> 5);
    if (row >= Nn) return;
    const int lane = threadIdx.x & 31;
    const float4 x1 = *(const float4*)(T1 + (size_t)row * DCH + lane * 4);
    const float4 x2 = *(const float4*)(T2 + (size_t)row * DCH + lane * 4);
    float m1, i1, m2, i2;
    row_stats(x1, m1, i1);
    row_stats(x2, m2, i2);
    const float xa[4] = {x1.x, x1.y, x1.z, x1.w};
    const float xb[4] = {x2.x, x2.y, x2.z, x2.w};
    #pragma unroll
    for (int c = 0; c < 4; ++c) {
        const int ch = lane * 4 + c;
        float y = (xa[c] - m1) * i1 * g1[ch] + b1[ch]
                + (xb[c] - m2) * i2 * g2[ch] + b2[ch];
        y = fmaxf(y, 0.f);
        feat[(size_t)row * DCH + ch]   = y;
        featbf[(size_t)row * DCH + ch] = (__bf16)y;
    }
}

// h = relu(gn(T,g,b)) -> bf16 only (feeds ctr2 GEMM)
__global__ __launch_bounds__(256) void gn_relu_to_bf16(
    const float* __restrict__ T, const float* __restrict__ g,
    const float* __restrict__ b, __bf16* __restrict__ out, int Nn)
{
    const int row = blockIdx.x * 8 + (threadIdx.x >> 5);
    if (row >= Nn) return;
    const int lane = threadIdx.x & 31;
    const float4 x = *(const float4*)(T + (size_t)row * DCH + lane * 4);
    float m, inv;
    row_stats(x, m, inv);
    const float xa[4] = {x.x, x.y, x.z, x.w};
    #pragma unroll
    for (int c = 0; c < 4; ++c) {
        const int ch = lane * 4 + c;
        float y = fmaxf((xa[c] - m) * inv * g[ch] + b[ch], 0.f);
        out[(size_t)row * DCH + ch] = (__bf16)y;
    }
}

// ---------------------------------------------------------------------------
// Scatter / gather
// ---------------------------------------------------------------------------
__global__ __launch_bounds__(256) void scatter_add_gather(
    float* __restrict__ tempo, const float* __restrict__ Z,
    const int* __restrict__ u, const int* __restrict__ v, int E)
{
    const int e = blockIdx.x * 8 + (threadIdx.x >> 5);
    if (e >= E) return;
    const int lane = threadIdx.x & 31;
    const float4 z = *(const float4*)(Z + (size_t)v[e] * DCH + lane * 4);
    float* dst = tempo + (size_t)u[e] * DCH + lane * 4;
    atomicAdd(dst + 0, z.x);
    atomicAdd(dst + 1, z.y);
    atomicAdd(dst + 2, z.z);
    atomicAdd(dst + 3, z.w);
}

__global__ __launch_bounds__(256) void scatter_add_direct(
    float* __restrict__ tempo, const float* __restrict__ Zs,
    const int* __restrict__ u, int E)
{
    const int e = blockIdx.x * 8 + (threadIdx.x >> 5);
    if (e >= E) return;
    const int lane = threadIdx.x & 31;
    const float4 z = *(const float4*)(Zs + (size_t)e * DCH + lane * 4);
    float* dst = tempo + (size_t)u[e] * DCH + lane * 4;
    atomicAdd(dst + 0, z.x);
    atomicAdd(dst + 1, z.y);
    atomicAdd(dst + 2, z.z);
    atomicAdd(dst + 3, z.w);
}

__global__ __launch_bounds__(256) void gather_rows_bf16(
    __bf16* __restrict__ X, const __bf16* __restrict__ featbf,
    const int* __restrict__ v, int E)
{
    const int e = blockIdx.x * 8 + (threadIdx.x >> 5);
    if (e >= E) return;
    const int lane = threadIdx.x & 31;
    ((unsigned long long*)(X + (size_t)e * DCH))[lane] =
        ((const unsigned long long*)(featbf + (size_t)v[e] * DCH))[lane];
}

// ---------------------------------------------------------------------------
// Orchestration
// ---------------------------------------------------------------------------
extern "C" void kernel_launch(void* const* d_in, const int* in_sizes, int n_in,
                              void* d_out, int out_size, void* d_ws, size_t ws_size,
                              hipStream_t stream)
{
    const float* ctrs   = (const float*)d_in[0];
    const float* feats  = (const float*)d_in[1];
    const float* w_in1  = (const float*)d_in[2];
    const float* b_in1  = (const float*)d_in[3];
    const float* w_in2  = (const float*)d_in[4];
    const float* g_in   = (const float*)d_in[5];
    const float* be_in  = (const float*)d_in[6];
    const float* w_seg1 = (const float*)d_in[7];
    const float* b_seg1 = (const float*)d_in[8];
    const float* w_seg2 = (const float*)d_in[9];
    const float* g_seg  = (const float*)d_in[10];
    const float* be_seg = (const float*)d_in[11];
    const float* ctr_w  = (const float*)d_in[12];
    const float* pre_w  = (const float*)d_in[13];
    const float* suc_w  = (const float*)d_in[14];
    const float* left_w = (const float*)d_in[15];
    const float* right_w= (const float*)d_in[16];
    const float* norm_g = (const float*)d_in[17];
    const float* norm_b = (const float*)d_in[18];
    const float* ctr2_w = (const float*)d_in[19];
    const float* ctr2_g = (const float*)d_in[20];
    const float* ctr2_b = (const float*)d_in[21];
    const int*   pre_u  = (const int*)d_in[22];
    const int*   pre_v  = (const int*)d_in[23];
    const int*   suc_u  = (const int*)d_in[24];
    const int*   suc_v  = (const int*)d_in[25];
    const int*   left_u = (const int*)d_in[26];
    const int*   left_v = (const int*)d_in[27];
    const int*   right_u= (const int*)d_in[28];
    const int*   right_v= (const int*)d_in[29];

    const int Nn = in_sizes[0] / 2;     // 200000
    const int E  = in_sizes[22] / 6;    // 200000
    const int E2 = in_sizes[26];        // 20000
    const int L = 4, S = 6;

    // ---- workspace carve ----
    auto align256 = [](size_t x) { return (x + 255) & ~(size_t)255; };
    char* ws = (char*)d_ws;
    size_t off = 0;
    __bf16* packedW = (__bf16*)(ws + off); off += align256((size_t)NMAT * MATELEMS * 2);
    __bf16* featbf  = (__bf16*)(ws + off); off += align256((size_t)Nn * DCH * 2);
    float*  tempo   = (float*) (ws + off); off += align256((size_t)Nn * DCH * 4);
    float*  Z       = (float*) (ws + off); off += align256((size_t)Nn * DCH * 4);
    __bf16* Xs      = (__bf16*)(ws + off); off += align256((size_t)E2 * DCH * 2);
    float*  Zs      = (float*) (ws + off); off += align256((size_t)E2 * DCH * 4);
    (void)ws_size;

    float* feat = (float*)d_out;        // f32 feat / residual lives in d_out

    auto pk = [&](int m) { return packedW + (size_t)m * MATELEMS; };

    const dim3 blk(256);
    const dim3 grdRow((Nn + 7) / 8);
    const dim3 grdGemmN((Nn + 127) / 128);
    const dim3 grdGemmE2((E2 + 127) / 128);
    const dim3 grdEdgeE((E + 7) / 8);
    const dim3 grdEdgeE2((E2 + 7) / 8);

    // 1) pack all 66 weight matrices to bf16 fragment order
    pack_weights<<<dim3(64, NMAT), blk, 0, stream>>>(
        w_in2, w_seg2, ctr_w, ctr2_w, left_w, right_w, pre_w, suc_w, packedW);

    // 2) encoders
    encoder_hidden<<<grdRow, blk, 0, stream>>>(ctrs, w_in1, b_in1, featbf, Nn);
    gemm_bf16_128<<<grdGemmN, blk, 0, stream>>>(featbf, pk(0), tempo, Nn);   // T1
    encoder_hidden<<<grdRow, blk, 0, stream>>>(feats, w_seg1, b_seg1, featbf, Nn);
    gemm_bf16_128<<<grdGemmN, blk, 0, stream>>>(featbf, pk(1), Z, Nn);       // T2
    encoder_combine<<<grdRow, blk, 0, stream>>>(tempo, Z, g_in, be_in, g_seg, be_seg,
                                                feat, featbf, Nn);

    // 3) fuse iterations
    for (int i = 0; i < L; ++i) {
        // temp = feat @ ctr_w[i]
        gemm_bf16_128<<<grdGemmN, blk, 0, stream>>>(featbf, pk(2 + i), tempo, Nn);

        for (int s = 0; s < S; ++s) {
            gemm_bf16_128<<<grdGemmN, blk, 0, stream>>>(featbf, pk(18 + i * 6 + s), Z, Nn);
            scatter_add_gather<<<grdEdgeE, blk, 0, stream>>>(
                tempo, Z, pre_u + (size_t)s * E, pre_v + (size_t)s * E, E);
            gemm_bf16_128<<<grdGemmN, blk, 0, stream>>>(featbf, pk(42 + i * 6 + s), Z, Nn);
            scatter_add_gather<<<grdEdgeE, blk, 0, stream>>>(
                tempo, Z, suc_u + (size_t)s * E, suc_v + (size_t)s * E, E);
        }

        // left / right: gather (E2 rows) -> GEMM -> scatter
        gather_rows_bf16<<<grdEdgeE2, blk, 0, stream>>>(Xs, featbf, left_v, E2);
        gemm_bf16_128<<<grdGemmE2, blk, 0, stream>>>(Xs, pk(10 + i), Zs, E2);
        scatter_add_direct<<<grdEdgeE2, blk, 0, stream>>>(tempo, Zs, left_u, E2);

        gather_rows_bf16<<<grdEdgeE2, blk, 0, stream>>>(Xs, featbf, right_v, E2);
        gemm_bf16_128<<<grdGemmE2, blk, 0, stream>>>(Xs, pk(14 + i), Zs, E2);
        scatter_add_direct<<<grdEdgeE2, blk, 0, stream>>>(tempo, Zs, right_u, E2);

        // h = relu(gn(temp)) (bf16) ; fused: feat = relu(gn(h@ctr2_w[i]) + res)
        gn_relu_to_bf16<<<grdRow, blk, 0, stream>>>(tempo, norm_g + (size_t)i * DCH,
                                                    norm_b + (size_t)i * DCH, featbf, Nn);
        gemm_bf16_128_gnres<<<grdGemmN, blk, 0, stream>>>(
            featbf, pk(6 + i), ctr2_g + (size_t)i * DCH, ctr2_b + (size_t)i * DCH,
            feat, featbf, Nn);
    }
}
